// _GridExtractor_61735859912710
// MI455X (gfx1250) — compile-verified, matrix operand order checked by reading, not run on hardware
//
#include <hip/hip_runtime.h>
#include <hip/hip_bf16.h>
#include <cstdint>

// ---------------------------------------------------------------------------
// GLCM contrast grid, factorized through per-pixel Gaussian moments.
//
//   S_m(x) = sum_j j^m * exp(-2 (x - c_j)^2),  m = 0,1,2
//   contrast(dir,d) = N / D with
//     N = sum_pairs S2(p)S0(q) + S0(p)S2(q) - 2 S1(p)S1(q)
//     D = sum_pairs S0(p)S0(q)
//
// Stage A computes the moment fields with v_wmma_f32_16x16x32_f16
// (16 pixels x 32 centers per WMMA, 8 K-chunks = 256 centers).
// Stage B computes the 16 shifted correlations. Stage C finalizes.
// ---------------------------------------------------------------------------

typedef __attribute__((ext_vector_type(16))) _Float16 v16h;
typedef __attribute__((ext_vector_type(8)))  float    v8f;

#define GLCM_HW   65536            // 256*256 pixels per image
#define GLCM_NB   4                // batch
#define FIELD_ELEMS ((size_t)GLCM_NB * GLCM_HW)
// ws float layout:
//   fields[6][NB*HW] : a-moments S0, S1/256, S2/65536  then b-moments (deg.)
//   acc[128]         : D,N per (combo,batch)
#define ACC_OFFSET (6 * FIELD_ELEMS)

// ---------------------------------------------------------------------------
// Stage A: per-pixel moments via WMMA.
// One wave handles 16 pixels. A-matrix: rows = pixels, K = 32 centers/chunk.
// Per the CDNA5 16-bit A layout, lane l owns row m = l&15; v16h element e maps
// to K = (e<8 ? e : e+8) + (l>=16 ? 8 : 0). B uses the mirrored layout with
// column n = l&15; only columns 0..2 are nonzero (weights 1, j/256, (j/256)^2).
// All inner-loop code is branchless so EXEC stays all-ones for the WMMAs.
// ---------------------------------------------------------------------------
__global__ void __launch_bounds__(256)
glcm_moments_wmma(const float* __restrict__ x, float* __restrict__ ws) {
  const int lane  = threadIdx.x & 31;
  const int wave  = threadIdx.x >> 5;
  const long tile = (long)blockIdx.x * 8 + wave;   // 16 pixels per wave-tile
  const long pixbase = tile * 16;                  // flat index into NB*HW
  const int  m     = lane & 15;                    // pixel row / weight column
  const int  khalf = lane >> 4;                    // which K half this lane owns

  const float xv = x[pixbase + m];

  // Per-lane moment-column selectors (hoisted v_cndmask, no control flow).
  const float is0 = (m == 0) ? 1.0f : 0.0f;
  const float is1 = (m == 1) ? 1.0f : 0.0f;
  const float is2 = (m == 2) ? 1.0f : 0.0f;

  const float step = 2.0f / 255.0f;

  v8f acc = {};
  #pragma unroll
  for (int kc = 0; kc < 8; ++kc) {
    v16h a, bw;
    const int jbase = kc * 32 + khalf * 8;         // this lane's K base
    #pragma unroll
    for (int e = 0; e < 16; ++e) {
      const int j = jbase + ((e < 8) ? e : (e + 8));  // center index 0..255
      // A: Gaussian soft-bin value for pixel m at center j
      const float c = fmaf((float)j, step, -1.0f);
      const float t = xv - c;
      a[e] = (_Float16)__expf(-2.0f * t * t);
      // B: moment weight for column m (scaled so f16 stays in range);
      // w = is0 + is1*(j/256) + is2*(j/256)^2, fully branchless.
      const float js = (float)j * (1.0f / 256.0f);
      bw[e] = (_Float16)fmaf(is2, js * js, fmaf(is1, js, is0));
    }
    acc = __builtin_amdgcn_wmma_f32_16x16x32_f16(
        /*neg_a=*/false, a, /*neg_b=*/false, bw,
        /*c_mod=*/(short)0, acc, /*reuse_a=*/false, /*reuse_b=*/false);
  }

  // C layout: lane l, vgpr r -> M = r + 8*(l>=16), N = l&15.
  // Columns 0..2 hold S0, S1/256, S2/65536 for the 16 pixels of this tile.
  if (m < 3) {
    float* f = ws + (size_t)m * FIELD_ELEMS;
    const long base = pixbase + khalf * 8;
    #pragma unroll
    for (int r = 0; r < 8; ++r) f[base + r] = acc[r];
  }

  // centers_b is linspace(-1,-1,256): every bin equals E = exp(-2 (x+1)^2),
  // so S0b = 256E, S1b/256 = 127.5E, S2b/65536 = (5559680/65536)E.
  if (lane < 16) {
    const float t = xv + 1.0f;
    const float E = __expf(-2.0f * t * t);
    ws[3 * FIELD_ELEMS + pixbase + m] = 256.0f * E;
    ws[4 * FIELD_ELEMS + pixbase + m] = 127.5f * E;
    ws[5 * FIELD_ELEMS + pixbase + m] = 84.833984375f * E;
  }
}

// ---------------------------------------------------------------------------
// Stage B: shifted correlations. One block per (combo, batch).
// combo = dir*4 + di; q = p + (dh, dw); validity == q in-bounds.
// ---------------------------------------------------------------------------
__global__ void __launch_bounds__(256)
glcm_corr(const float* __restrict__ ws, float* __restrict__ acc) {
  const int combo = blockIdx.x >> 2;   // 0..15
  const int b     = blockIdx.x & 3;    // batch
  const int dir   = combo >> 2;
  const int di    = combo & 3;

  const int d_odd  = 1 + 2 * di;                 // 1,3,5,7   (dir 0 and 90)
  const int d_even = (di == 0) ? 1 : 2 * di;     // 1,2,4,6   (dir 45 and 135)
  int dh, dw;
  if (dir == 0)      { dh = 0;       dw =  d_odd;  }
  else if (dir == 1) { dh = -d_even; dw =  d_even; }
  else if (dir == 2) { dh = -d_odd;  dw =  0;      }
  else               { dh = -d_even; dw = -d_even; }

  const size_t foff = (combo < 8) ? 0 : 3 * FIELD_ELEMS;
  const float* m0 = ws + foff + (size_t)b * GLCM_HW;
  const float* m1 = m0 + FIELD_ELEMS;
  const float* m2 = m1 + FIELD_ELEMS;

  float D = 0.0f, N = 0.0f;
  for (int p = threadIdx.x; p < GLCM_HW; p += 256) {
    __builtin_prefetch(m0 + p + 4096, 0, 0);     // global_prefetch_b8, speculative
    const int h = p >> 8, w = p & 255;
    const int qh = h + dh, qw = w + dw;
    if ((unsigned)qh < 256u && (unsigned)qw < 256u) {
      const int q = (qh << 8) | qw;
      const float p0 = m0[p], p1 = m1[p], p2 = m2[p];
      const float q0 = m0[q], q1 = m1[q], q2 = m2[q];
      D = fmaf(p0, q0, D);
      N += fmaf(p2, q0, fmaf(p0, q2, -2.0f * p1 * q1));
    }
  }

  __shared__ float sD[256], sN[256];
  sD[threadIdx.x] = D;
  sN[threadIdx.x] = N;
  __syncthreads();
  for (int s = 128; s > 0; s >>= 1) {
    if ((int)threadIdx.x < s) {
      sD[threadIdx.x] += sD[threadIdx.x + s];
      sN[threadIdx.x] += sN[threadIdx.x + s];
    }
    __syncthreads();
  }
  if (threadIdx.x == 0) {
    acc[2 * blockIdx.x]     = sD[0];
    acc[2 * blockIdx.x + 1] = sN[0];
  }
}

// ---------------------------------------------------------------------------
// Stage C: finalize. out flat index f = bo*16 + i*4 + j maps to
// grid_idx = f/4, batch = f%4 (concatenate over axis 0 then reshape).
// Moments were stored scaled by 1/256 and 1/65536 -> restore with *65536.
// ---------------------------------------------------------------------------
__global__ void glcm_finalize(const float* __restrict__ acc, float* __restrict__ out) {
  const int f = threadIdx.x;           // 0..63
  const int grid_idx = f >> 2;
  const int b = f & 3;
  const float D = acc[2 * (grid_idx * 4 + b)];
  const float N = acc[2 * (grid_idx * 4 + b) + 1];
  const float denom = (D == 0.0f) ? 1.0f : D;
  out[f] = 65536.0f * N / denom;
}

extern "C" void kernel_launch(void* const* d_in, const int* in_sizes, int n_in,
                              void* d_out, int out_size, void* d_ws, size_t ws_size,
                              hipStream_t stream) {
  const float* x = (const float*)d_in[0];       // (4,1,256,256) fp32
  float* ws  = (float*)d_ws;
  float* out = (float*)d_out;                   // 64 floats
  float* acc = ws + ACC_OFFSET;

  // Stage A: NB*HW/16 = 16384 wave-tiles, 8 waves (256 threads) per block.
  glcm_moments_wmma<<<2048, 256, 0, stream>>>(x, ws);
  // Stage B: one block per (combo, batch) = 64 blocks.
  glcm_corr<<<64, 256, 0, stream>>>(ws, acc);
  // Stage C: single tiny block.
  glcm_finalize<<<1, 64, 0, stream>>>(acc, out);
}
